// Model_89318139887923
// MI455X (gfx1250) — matrix operand order repaired and checked by reading, not compiled, not run on hardware
//
#include <hip/hip_runtime.h>
#include <hip/hip_bf16.h>

#define SEQ   2048
#define BATCH 64
#define HID   512
#define DEPTH 2

typedef __attribute__((ext_vector_type(16))) __bf16  v16bf;
typedef __attribute__((ext_vector_type(8)))  float   v8f;
typedef __attribute__((ext_vector_type(4)))  unsigned int u32x4;
typedef __attribute__((ext_vector_type(2)))  unsigned int u32x2;

union FragBF {
    v16bf v;
    u32x4 q[2];
};

__device__ __forceinline__ unsigned short f2bfu(float x) {
    unsigned u = __builtin_bit_cast(unsigned, x);
    unsigned r = (u + 0x7FFFu + ((u >> 16) & 1u)) >> 16;
    return (unsigned short)r;
}
__device__ __forceinline__ __bf16 f2bf(float x) {
    unsigned short s = f2bfu(x);
    return __builtin_bit_cast(__bf16, s);
}
__device__ __forceinline__ float sigm(float x) {
    return 1.0f / (1.0f + __expf(-x));
}

// ---------------------------------------------------------------------------
// Weight prep: fp32 -> bf16 for Wih/Whh, and bsum = bih + bhh (fp32).
// ---------------------------------------------------------------------------
__global__ void prep_kernel(const float* __restrict__ Wih, const float* __restrict__ Whh,
                            const float* __restrict__ bih, const float* __restrict__ bhh,
                            __bf16* __restrict__ Wihb, __bf16* __restrict__ Whhb,
                            float* __restrict__ bsum, int nW, int nB) {
    int e = blockIdx.x * 256 + threadIdx.x;
    if (e < nW) {
        Wihb[e] = f2bf(Wih[e]);
        Whhb[e] = f2bf(Whh[e]);
    }
    if (e < nB) bsum[e] = bih[e] + bhh[e];
}

// ---------------------------------------------------------------------------
// One LSTM timestep for one layer.
// Grid: 128 blocks = 4 batch-tiles x 32 H-tiles. Block = 128 threads = 4 waves,
// wave w computes gate w's 16x16 tile via v_wmma_f32_16x16x32_bf16.
// The Wih and Whh GEMMs are interleaved in one K-loop with two independent
// accumulator chains so the matrix pipe never stalls on a serial WMMA chain.
//   xin   : [B,H] bf16 input at time t, or null -> fuse embedding lookup
//   hprev : [B,H] bf16 h_{t-1}, or null -> t==0 (h=0, c=0)
//   hout  : [B,H] bf16 h_t destination
// flags bit0: also write fp32 h into h32 (for final LayerNorm).
// ---------------------------------------------------------------------------
__global__ __launch_bounds__(128)
void lstm_step_kernel(const int* __restrict__ inp, const float* __restrict__ embed,
                      const __bf16* __restrict__ xin, const __bf16* __restrict__ hprev,
                      __bf16* __restrict__ hout,
                      const __bf16* __restrict__ Wihb, const __bf16* __restrict__ Whhb,
                      const float* __restrict__ bsum,
                      float* __restrict__ Cbuf, float* __restrict__ h32,
                      int t, int flags) {
    __shared__ __align__(16) unsigned char smem[32768];
    __bf16* xs = (__bf16*)smem;              // [16][512] x_t rows
    __bf16* hs = (__bf16*)(smem + 16384);    // [16][512] h_{t-1} rows

    const int tid = threadIdx.x;
    const int m0  = (blockIdx.x >> 5) << 4;  // batch tile base
    const int n0  = (blockIdx.x & 31) << 4;  // H column tile base

    // ---- Stage x_t (16 rows x 512) into LDS ----
    if (xin) {
        u32x4* xd = (u32x4*)xs;
#pragma unroll
        for (int i = 0; i < 8; ++i) {
            int c = tid + i * 128;           // 1024 chunks of 16B
            int row = c >> 6;
            int e8  = (c & 63) * 8;
            xd[c] = *(const u32x4*)(xin + (size_t)(m0 + row) * HID + e8);
        }
    } else {
        // fused embedding: x = embed[idx0] + embed[idx1], per row
#pragma unroll
        for (int row = 0; row < 16; ++row) {
            int b  = m0 + row;
            int i0 = inp[b * SEQ + t];
            int i1 = inp[(BATCH + b) * SEQ + t];
            const float4 a = *(const float4*)(embed + i0 * HID + tid * 4);
            const float4 d = *(const float4*)(embed + i1 * HID + tid * 4);
            u32x2 p;
            p.x = (unsigned)f2bfu(a.x + d.x) | ((unsigned)f2bfu(a.y + d.y) << 16);
            p.y = (unsigned)f2bfu(a.z + d.z) | ((unsigned)f2bfu(a.w + d.w) << 16);
            *(u32x2*)(xs + row * HID + tid * 4) = p;
        }
    }
    // ---- Stage h_{t-1} into LDS ----
    if (hprev) {
        u32x4* hd = (u32x4*)hs;
#pragma unroll
        for (int i = 0; i < 8; ++i) {
            int c = tid + i * 128;
            int row = c >> 6;
            int e8  = (c & 63) * 8;
            hd[c] = *(const u32x4*)(hprev + (size_t)(m0 + row) * HID + e8);
        }
    }
    __syncthreads();

    // ---- WMMA GEMM: gates[16b x 16n] for this wave's gate ----
    const int lane = tid & 31;
    const int wv   = tid >> 5;        // gate index 0..3 (i,f,g,o)
    const int half = lane >> 4;
    const int lm   = lane & 15;

    v8f acc0, acc1;
    float bv = bsum[wv * HID + n0 + lm];
#pragma unroll
    for (int r = 0; r < 8; ++r) { acc0[r] = bv; acc1[r] = 0.0f; }

    const __bf16* wr   = Wihb + (size_t)(wv * HID + n0 + lm) * HID;  // weight row (B operand)
    const __bf16* wrh  = Whhb + (size_t)(wv * HID + n0 + lm) * HID;
    const __bf16* xrow = xs + lm * HID;                              // activation row (A operand)

    if (hprev) {
        const __bf16* hrow = hs + lm * HID;
        FragBF a0, b0, a1, b1;
#pragma unroll 4
        for (int kc = 0; kc < HID; kc += 32) {
            // A 16-bit 16x32 layout: lanes m, VGPR0-3 K=half*8+[0..8), VGPR4-7 K=16+half*8+[0..8)
            a0.q[0] = *(const u32x4*)(xrow + kc + half * 8);
            a0.q[1] = *(const u32x4*)(xrow + kc + 16 + half * 8);
            // B 16-bit 32x16 layout: lane n, VGPR0-3 K=half*8+[0..8), VGPR4-7 K=16+half*8+[0..8)
            b0.q[0] = *(const u32x4*)(wr + kc + half * 8);
            b0.q[1] = *(const u32x4*)(wr + kc + 16 + half * 8);
            a1.q[0] = *(const u32x4*)(hrow + kc + half * 8);
            a1.q[1] = *(const u32x4*)(hrow + kc + 16 + half * 8);
            b1.q[0] = *(const u32x4*)(wrh + kc + half * 8);
            b1.q[1] = *(const u32x4*)(wrh + kc + 16 + half * 8);
            acc0 = __builtin_amdgcn_wmma_f32_16x16x32_bf16(false, a0.v, false, b0.v,
                                                           (short)0, acc0, false, false);
            acc1 = __builtin_amdgcn_wmma_f32_16x16x32_bf16(false, a1.v, false, b1.v,
                                                           (short)0, acc1, false, false);
        }
        acc0 = acc0 + acc1;
    } else {
        FragBF a, b;
#pragma unroll 4
        for (int kc = 0; kc < HID; kc += 32) {
            a.q[0] = *(const u32x4*)(xrow + kc + half * 8);
            a.q[1] = *(const u32x4*)(xrow + kc + 16 + half * 8);
            b.q[0] = *(const u32x4*)(wr + kc + half * 8);
            b.q[1] = *(const u32x4*)(wr + kc + 16 + half * 8);
            acc0 = __builtin_amdgcn_wmma_f32_16x16x32_bf16(false, a.v, false, b.v,
                                                           (short)0, acc0, false, false);
        }
    }

    // ---- Gate exchange through LDS, then fused LSTM cell update ----
    __syncthreads();
    float* gl = (float*)smem;          // [4][16][16] fp32 gate tiles (4KB)
#pragma unroll
    for (int r = 0; r < 8; ++r)        // C/D layout: m = half*8 + r, n = lane%16
        gl[wv * 256 + (half * 8 + r) * 16 + lm] = acc0[r];
    __syncthreads();

#pragma unroll
    for (int i = 0; i < 2; ++i) {
        int e = tid + i * 128;         // element (m,n) of the 16x16 tile
        int m = e >> 4, n = e & 15;
        float iv = gl[e];
        float fv = gl[256 + e];
        float gv = gl[512 + e];
        float ov = gl[768 + e];
        int gi = (m0 + m) * HID + (n0 + n);
        float cp = hprev ? Cbuf[gi] : 0.0f;
        float cn = sigm(fv) * cp + sigm(iv) * tanhf(gv);
        float hv = sigm(ov) * tanhf(cn);
        Cbuf[gi] = cn;
        hout[gi] = f2bf(hv);
        if (flags & 1) h32[gi] = hv;
    }
}

// ---------------------------------------------------------------------------
// Final: LayerNorm of last h + projection. One block per batch row.
// ---------------------------------------------------------------------------
__global__ __launch_bounds__(128)
void final_kernel(const float* __restrict__ h32, const float* __restrict__ g,
                  const float* __restrict__ be, const float* __restrict__ Wp,
                  const float* __restrict__ bp, float* __restrict__ out) {
    __shared__ float s1[128], s2[128];
    int b = blockIdx.x, tid = threadIdx.x;
    const float* hr = h32 + b * HID;
    float s = 0.f, sq = 0.f;
    for (int h = tid; h < HID; h += 128) { float v = hr[h]; s += v; sq += v * v; }
    s1[tid] = s; s2[tid] = sq; __syncthreads();
    for (int st = 64; st > 0; st >>= 1) {
        if (tid < st) { s1[tid] += s1[tid + st]; s2[tid] += s2[tid + st]; }
        __syncthreads();
    }
    float mu  = s1[0] * (1.0f / HID);
    float var = s2[0] * (1.0f / HID) - mu * mu;
    float rs  = rsqrtf(var + 1e-5f);
    __syncthreads();
    float p = 0.f;
    for (int h = tid; h < HID; h += 128)
        p += ((hr[h] - mu) * rs * g[h] + be[h]) * Wp[h];
    s1[tid] = p; __syncthreads();
    for (int st = 64; st > 0; st >>= 1) {
        if (tid < st) s1[tid] += s1[tid + st];
        __syncthreads();
    }
    if (tid == 0) out[b] = s1[0] + bp[0];
}

// ---------------------------------------------------------------------------
extern "C" void kernel_launch(void* const* d_in, const int* in_sizes, int n_in,
                              void* d_out, int out_size, void* d_ws, size_t ws_size,
                              hipStream_t stream) {
    (void)in_sizes; (void)n_in; (void)out_size; (void)ws_size;
    const int*   inp   = (const int*)d_in[0];
    const float* embed = (const float*)d_in[1];
    const float* Wih   = (const float*)d_in[2];
    const float* Whh   = (const float*)d_in[3];
    const float* bih   = (const float*)d_in[4];
    const float* bhh   = (const float*)d_in[5];
    const float* ln_g  = (const float*)d_in[6];
    const float* ln_b  = (const float*)d_in[7];
    const float* Wp    = (const float*)d_in[8];
    const float* bp    = (const float*)d_in[9];

    const size_t BH   = (size_t)BATCH * HID;
    const size_t WSZ  = (size_t)DEPTH * 4 * HID * HID;   // elements per weight tensor

    char* ws = (char*)d_ws;
    __bf16* X1   = (__bf16*)ws;                 ws += (size_t)SEQ * BH * 2;   // layer-0 output seq
    __bf16* Wihb = (__bf16*)ws;                 ws += WSZ * 2;
    __bf16* Whhb = (__bf16*)ws;                 ws += WSZ * 2;
    float*  bsum = (float*)ws;                  ws += (size_t)DEPTH * 4 * HID * 4;
    float*  Cbuf = (float*)ws;                  ws += BH * 4;
    float*  h32  = (float*)ws;                  ws += BH * 4;
    __bf16* hbA  = (__bf16*)ws;                 ws += BH * 2;
    __bf16* hbB  = (__bf16*)ws;

    // 1) weights fp32 -> bf16, bias sum
    prep_kernel<<<(int)((WSZ + 255) / 256), 256, 0, stream>>>(
        Wih, Whh, bih, bhh, Wihb, Whhb, bsum, (int)WSZ, DEPTH * 4 * HID);

    // 2) layer 0: fused embedding input, writes full output sequence X1
    for (int t = 0; t < SEQ; ++t) {
        lstm_step_kernel<<<128, 128, 0, stream>>>(
            inp, embed, /*xin=*/nullptr,
            t ? (X1 + (size_t)(t - 1) * BH) : nullptr,
            X1 + (size_t)t * BH,
            Wihb, Whhb, bsum, Cbuf, h32, t, 0);
    }
    // 3) layer 1: reads X1, ping-pong h buffers, last step stores fp32 h
    for (int t = 0; t < SEQ; ++t) {
        lstm_step_kernel<<<128, 128, 0, stream>>>(
            nullptr, nullptr, X1 + (size_t)t * BH,
            t ? ((t & 1) ? hbA : hbB) : nullptr,
            (t & 1) ? hbB : hbA,
            Wihb + (size_t)4 * HID * HID, Whhb + (size_t)4 * HID * HID,
            bsum + 4 * HID, Cbuf, h32, t, (t == SEQ - 1) ? 1 : 0);
    }
    // 4) LayerNorm + projection
    final_kernel<<<BATCH, 128, 0, stream>>>(h32, ln_g, ln_b, Wp, bp, (float*)d_out);
}
